// FeatureRefiner_12979391168962
// MI455X (gfx1250) — matrix operand order repaired and checked by reading, not compiled, not run on hardware
//
#include <hip/hip_runtime.h>

#define F     1024
#define NTAR  60000
#define NMSG  45000

#define BM 64
#define BN 128
#define BK 32
#define LDA 40   // bf16 elems per LDS row (32 + 8 pad) -> 80B row stride
#define LDB 40

typedef __attribute__((ext_vector_type(16))) __bf16       v16bf;
typedef __attribute__((ext_vector_type(8)))  float        v8f;
typedef __attribute__((ext_vector_type(4)))  float        f32x4;
typedef __attribute__((ext_vector_type(4)))  unsigned int u32x4;
typedef __attribute__((ext_vector_type(2)))  unsigned int u32x2;

union FragU { u32x4 q[2]; v16bf v; };

__device__ __forceinline__ unsigned short f2bf(float f) {
  unsigned int u = __float_as_uint(f);
  u += 0x7FFFu + ((u >> 16) & 1u);      // round-to-nearest-even
  return (unsigned short)(u >> 16);
}

__device__ __forceinline__ void load8(const float* p, float* r) {
  f32x4 a = *reinterpret_cast<const f32x4*>(p);
  f32x4 b = *reinterpret_cast<const f32x4*>(p + 4);
#pragma unroll
  for (int i = 0; i < 4; ++i) { r[i] = a[i]; r[4 + i] = b[i]; }
}

// Async copy 32 bytes global -> LDS per lane (2 x b128, ASYNCcnt-tracked).
// INST_OFFSET is added to BOTH the LDS and global addresses (ISA 15.18.3),
// so one base covers both 16B chunks.
__device__ __forceinline__ void async_copy32(unsigned lds_off, const void* gaddr) {
  asm volatile("global_load_async_to_lds_b128 %0, %1, off"
               :: "v"(lds_off), "v"(gaddr) : "memory");
  asm volatile("global_load_async_to_lds_b128 %0, %1, off offset:16"
               :: "v"(lds_off), "v"(gaddr) : "memory");
}

// ---------------- prep kernels ----------------

__global__ void prep_weights(const float* __restrict__ w1, const float* __restrict__ w2,
                             unsigned short* __restrict__ w1bf, unsigned short* __restrict__ w2bf) {
  size_t i = ((size_t)blockIdx.x * blockDim.x + threadIdx.x) * 4;
  if (i >= (size_t)F * F) return;
  f32x4 a = *reinterpret_cast<const f32x4*>(w1 + i);
  f32x4 b = *reinterpret_cast<const f32x4*>(w2 + i);
  u32x2 pa, pb;
  pa[0] = (unsigned)f2bf(a[0]) | ((unsigned)f2bf(a[1]) << 16);
  pa[1] = (unsigned)f2bf(a[2]) | ((unsigned)f2bf(a[3]) << 16);
  pb[0] = (unsigned)f2bf(b[0]) | ((unsigned)f2bf(b[1]) << 16);
  pb[1] = (unsigned)f2bf(b[2]) | ((unsigned)f2bf(b[3]) << 16);
  *reinterpret_cast<u32x2*>(w1bf + i) = pa;
  *reinterpret_cast<u32x2*>(w2bf + i) = pb;
}

__global__ void init_inv(int* __restrict__ inv_o, int* __restrict__ inv_s) {
  int i = blockIdx.x * blockDim.x + threadIdx.x;
  if (i < NTAR) { inv_o[i] = -1; inv_s[i] = -1; }
}

__global__ void scatter_inv(const int* __restrict__ o_idx, const int* __restrict__ s_idx,
                            int* __restrict__ inv_o, int* __restrict__ inv_s) {
  int i = blockIdx.x * blockDim.x + threadIdx.x;
  if (i < NMSG) { inv_o[o_idx[i]] = i; inv_s[s_idx[i]] = i; }
}

// ---------------- fused GEMM kernel ----------------

__launch_bounds__(256)
__global__ void refiner_gemm(const float* __restrict__ msg_o, const float* __restrict__ msg_s,
                             const float* __restrict__ tar,
                             const unsigned short* __restrict__ w1bf,
                             const unsigned short* __restrict__ w2bf,
                             const float* __restrict__ bias1, const float* __restrict__ bias2,
                             const int* __restrict__ inv_o, const int* __restrict__ inv_s,
                             float* __restrict__ out) {
  __shared__ __align__(16) unsigned short sS[BM * LDA];
  __shared__ __align__(16) unsigned short sT[BM * LDA];
  __shared__ __align__(16) unsigned short sW1[2][BN * LDB];   // double-buffered (async)
  __shared__ __align__(16) unsigned short sW2[2][BN * LDB];

  const int tid  = threadIdx.x;
  const int lane = tid & 31;
  const int lr   = lane & 15;      // row/col within 16-wide stripe
  const int lhi  = lane >> 4;      // which half-wave
  const int wv   = tid >> 5;       // wave id 0..7
  const int wm   = wv & 1;         // 2 M-groups of 32 rows
  const int wn   = wv >> 1;        // 4 N-groups of 32 cols

  const int n0 = blockIdx.x * BN;
  const int m0 = blockIdx.y * BM;

  // staging assignments
  const int aRow   = tid >> 2;          // 0..63
  const int aChunk = (tid & 3) * 8;     // k-offset 0,8,16,24 (8 floats each)
  const int wRow   = tid >> 1;          // 0..127
  const int wHalf  = (tid & 1) * 16;    // k-offset 0 or 16 (16 bf16 each)

  const int  mA = m0 + aRow;
  const bool mAvalid = (mA < NTAR);
  int ioA = -1, isA = -1;
  if (mAvalid) { ioA = inv_o[mA]; isA = inv_s[mA]; }
  const float* poA = msg_o + (size_t)(ioA < 0 ? 0 : ioA) * F;
  const float* psA = msg_s + (size_t)(isA < 0 ? 0 : isA) * F;
  const float* ptA = tar   + (size_t)(mAvalid ? mA : 0) * F;

  const unsigned short* gW1 = w1bf + (size_t)(n0 + wRow) * F + wHalf;
  const unsigned short* gW2 = w2bf + (size_t)(n0 + wRow) * F + wHalf;
  const unsigned lW1[2] = {(unsigned)(size_t)&sW1[0][wRow * LDB + wHalf],
                           (unsigned)(size_t)&sW1[1][wRow * LDB + wHalf]};
  const unsigned lW2[2] = {(unsigned)(size_t)&sW2[0][wRow * LDB + wHalf],
                           (unsigned)(size_t)&sW2[1][wRow * LDB + wHalf]};

  v8f accS[2][2], accT[2][2];
#pragma unroll
  for (int a = 0; a < 2; ++a)
#pragma unroll
    for (int b = 0; b < 2; ++b) {
      v8f z = {0.f, 0.f, 0.f, 0.f, 0.f, 0.f, 0.f, 0.f};
      accS[a][b] = z;
      accT[a][b] = z;
    }

  // prologue: async-fetch W tiles for k0 = 0 into buffer 0
  async_copy32(lW1[0], gW1);
  async_copy32(lW2[0], gW2);

  int parity = 0;
  for (int k0 = 0; k0 < F; k0 += BK) {
    __syncthreads();  // all waves done reading S/T and W[parity^1] of prev iter

    const int kn = (k0 + BK < F) ? (k0 + BK) : 0;   // uniform: wrap keeps waits simple

    // ---- issue async W1/W2 tiles for NEXT iteration into buffer parity^1 ----
    async_copy32(lW1[parity ^ 1], gW1 + kn);
    async_copy32(lW2[parity ^ 1], gW2 + kn);

    // ---- stage S (= 0.5*(scatter_o + scatter_s)) and T tiles, fp32 -> bf16 ----
    {
      float vs[8], vt[8];
#pragma unroll
      for (int i = 0; i < 8; ++i) { vs[i] = 0.f; vt[i] = 0.f; }
      if (mAvalid) {
        if (ioA >= 0) {
          float r[8]; load8(poA + k0 + aChunk, r);
#pragma unroll
          for (int i = 0; i < 8; ++i) vs[i] += r[i];
          __builtin_prefetch(poA + kn + aChunk, 0, 0);
        }
        if (isA >= 0) {
          float r[8]; load8(psA + k0 + aChunk, r);
#pragma unroll
          for (int i = 0; i < 8; ++i) vs[i] += r[i];
          __builtin_prefetch(psA + kn + aChunk, 0, 0);
        }
        load8(ptA + k0 + aChunk, vt);
        __builtin_prefetch(ptA + kn + aChunk, 0, 0);
      }
      u32x4 ps, pt;
#pragma unroll
      for (int i = 0; i < 4; ++i) {
        ps[i] = (unsigned)f2bf(0.5f * vs[2 * i]) | ((unsigned)f2bf(0.5f * vs[2 * i + 1]) << 16);
        pt[i] = (unsigned)f2bf(vt[2 * i])        | ((unsigned)f2bf(vt[2 * i + 1]) << 16);
      }
      *reinterpret_cast<u32x4*>(&sS[aRow * LDA + aChunk]) = ps;
      *reinterpret_cast<u32x4*>(&sT[aRow * LDA + aChunk]) = pt;
    }

    // oldest 4 async ops (current buffer's W tiles) must have landed;
    // the 4 just issued for the next buffer may stay in flight.
    asm volatile("s_wait_asynccnt 0x4" ::: "memory");
    __syncthreads();

    // ---- load fragments (per ISA wave32 layouts) and issue WMMAs ----
    FragU aS[2], aT[2], bW1[2], bW2[2];
#pragma unroll
    for (int tr = 0; tr < 2; ++tr) {
      // A 16x32 bf16: lane<16 -> K{0..7,16..23}; lane>=16 -> K{8..15,24..31}
      const char* pS = reinterpret_cast<const char*>(&sS[(wm * 32 + tr * 16 + lr) * LDA]);
      const char* pT = reinterpret_cast<const char*>(&sT[(wm * 32 + tr * 16 + lr) * LDA]);
      aS[tr].q[0] = *reinterpret_cast<const u32x4*>(pS + lhi * 16);
      aS[tr].q[1] = *reinterpret_cast<const u32x4*>(pS + 32 + lhi * 16);
      aT[tr].q[0] = *reinterpret_cast<const u32x4*>(pT + lhi * 16);
      aT[tr].q[1] = *reinterpret_cast<const u32x4*>(pT + 32 + lhi * 16);
    }
#pragma unroll
    for (int tc = 0; tc < 2; ++tc) {
      // B 32x16 bf16: lane n holds col n; lane<16 -> K{0..7,16..23}; lane>=16 -> K{8..15,24..31}
      // B[k][n] = W[n0+col][k] => contiguous run of a W row
      const char* p1 = reinterpret_cast<const char*>(&sW1[parity][(wn * 32 + tc * 16 + lr) * LDB]);
      const char* p2 = reinterpret_cast<const char*>(&sW2[parity][(wn * 32 + tc * 16 + lr) * LDB]);
      bW1[tc].q[0] = *reinterpret_cast<const u32x4*>(p1 + lhi * 16);
      bW1[tc].q[1] = *reinterpret_cast<const u32x4*>(p1 + 32 + lhi * 16);
      bW2[tc].q[0] = *reinterpret_cast<const u32x4*>(p2 + lhi * 16);
      bW2[tc].q[1] = *reinterpret_cast<const u32x4*>(p2 + 32 + lhi * 16);
    }

#pragma unroll
    for (int tr = 0; tr < 2; ++tr)
#pragma unroll
      for (int tc = 0; tc < 2; ++tc) {
        accS[tr][tc] = __builtin_amdgcn_wmma_f32_16x16x32_bf16(
            false, aS[tr].v, false, bW1[tc].v, (short)0, accS[tr][tc], false, false);
        accT[tr][tc] = __builtin_amdgcn_wmma_f32_16x16x32_bf16(
            false, aT[tr].v, false, bW2[tc].v, (short)0, accT[tr][tc], false, false);
      }

    parity ^= 1;
  }

  // ---- epilogue: out = summarized + relu(accS + b1) + relu(accT + b2) ----
  // C/D layout: VGPR v -> M = v + 8*lhi ; N = lane%16
#pragma unroll
  for (int tr = 0; tr < 2; ++tr) {
#pragma unroll
    for (int v = 0; v < 8; ++v) {
      int m = m0 + wm * 32 + tr * 16 + lhi * 8 + v;
      if (m >= NTAR) continue;
      int io = inv_o[m];
      int is = inv_s[m];
#pragma unroll
      for (int tc = 0; tc < 2; ++tc) {
        int n = n0 + wn * 32 + tc * 16 + lr;
        float sv = 0.f;
        if (io >= 0) sv += msg_o[(size_t)io * F + n];
        if (is >= 0) sv += msg_s[(size_t)is * F + n];
        sv *= 0.5f;
        float h1 = accS[tr][tc][v] + bias1[n];
        float h2 = accT[tr][tc][v] + bias2[n];
        h1 = h1 > 0.f ? h1 : 0.f;
        h2 = h2 > 0.f ? h2 : 0.f;
        out[(size_t)m * F + n] = sv + h1 + h2;
      }
    }
  }
}

// ---------------- host launcher ----------------

extern "C" void kernel_launch(void* const* d_in, const int* in_sizes, int n_in,
                              void* d_out, int out_size, void* d_ws, size_t ws_size,
                              hipStream_t stream) {
  const float* msg_o = (const float*)d_in[0];
  const float* msg_s = (const float*)d_in[1];
  const int*   o_idx = (const int*)d_in[2];
  const int*   s_idx = (const int*)d_in[3];
  const float* tar   = (const float*)d_in[4];
  const float* w1    = (const float*)d_in[5];
  const float* b1    = (const float*)d_in[6];
  const float* w2    = (const float*)d_in[7];
  const float* b2    = (const float*)d_in[8];
  float*       out   = (float*)d_out;

  char* ws = (char*)d_ws;
  unsigned short* w1bf = (unsigned short*)ws;
  unsigned short* w2bf = w1bf + (size_t)F * F;
  int* inv_o = (int*)(ws + 2 * (size_t)F * F * sizeof(unsigned short));
  int* inv_s = inv_o + NTAR;

  prep_weights<<<dim3((F * F / 4 + 255) / 256), dim3(256), 0, stream>>>(w1, w2, w1bf, w2bf);
  init_inv<<<dim3((NTAR + 255) / 256), dim3(256), 0, stream>>>(inv_o, inv_s);
  scatter_inv<<<dim3((NMSG + 255) / 256), dim3(256), 0, stream>>>(o_idx, s_idx, inv_o, inv_s);

  refiner_gemm<<<dim3(F / BN, (NTAR + BM - 1) / BM), dim3(256), 0, stream>>>(
      msg_o, msg_s, tar, w1bf, w2bf, b1, b2, inv_o, inv_s, out);
}
